// RobustCaponFrequencyFusion_76965813944523
// MI455X (gfx1250) — matrix operand order repaired and checked by reading, not compiled, not run on hardware
//
#include <hip/hip_runtime.h>
#include <math.h>

typedef __attribute__((ext_vector_type(16))) _Float16 v16h;
typedef __attribute__((ext_vector_type(8)))  float    v8f;

#define Hdim 128
#define Wdim 128
#define HWn  (Hdim * Wdim)      // 16384
#define Cdim 64
#define Kstr 8                  // streams
#define Bdim 2
#define Cmid 32

// ---------------------------------------------------------------------------
// Kernel 1: fused 1x1 conv -> BN(inference) -> ReLU -> 1x1 conv, via WMMA.
// GEMM: h[32 x P] = w1[32x64] * X[64 x P], P = B*K*H*W = 262144 pixels.
// One wave handles 16 pixels x 32 channels with 4x v_wmma_f32_16x16x32_f16.
// ---------------------------------------------------------------------------
__global__ __launch_bounds__(256)
void conv_sensor_kernel(const float* __restrict__ X,
                        const float* __restrict__ w1,
                        const float* __restrict__ bn_scale,
                        const float* __restrict__ bn_bias,
                        const float* __restrict__ bn_mean,
                        const float* __restrict__ bn_var,
                        const float* __restrict__ w2,
                        float* __restrict__ S)
{
    const int lane  = threadIdx.x & 31;
    const int wv    = threadIdx.x >> 5;
    const int chunk = blockIdx.x * 8 + wv;   // 16384 chunks total, all full
    const int g     = chunk >> 10;           // image index in [0, B*K)
    const int q0    = (chunk & 1023) << 4;   // pixel base within image
    const int n     = lane & 15;             // pixel column within tile
    const int half  = lane >> 4;             // lane-group (K-half select)

    // ---- A tiles (w1, f32 -> f16), ISA 16-bit A 16x32 layout:
    // lane l holds row m = l&15; VGPRs 0-3 pack K = half*8 + {0..7},
    // VGPRs 4-7 pack K = 16 + half*8 + {0..7}.
    v16h a00, a01, a10, a11;
    {
        const int m0 = lane & 15;
        #pragma unroll
        for (int i = 0; i < 16; ++i) {
            const int kl = (i < 8) ? (half * 8 + i) : (16 + half * 8 + (i - 8));
            a00[i] = (_Float16)w1[(m0     ) * Cdim +      kl];
            a01[i] = (_Float16)w1[(m0     ) * Cdim + 32 + kl];
            a10[i] = (_Float16)w1[(m0 + 16) * Cdim +      kl];
            a11[i] = (_Float16)w1[(m0 + 16) * Cdim + 32 + kl];
        }
    }

    // ---- B tiles (X column tile, f32 -> f16), ISA 16-bit B 32x16 layout:
    // lane l holds column n = l&15; elements 0-7 pack K = half*8 + {0..7}, elements 8-15 pack K = 16 + half*8 + {0..7}.
    const float* Xg = X + (size_t)g * Cdim * HWn + q0 + n;
    v16h b0, b1;
    #pragma unroll
    for (int i = 0; i < 16; ++i) {
        const int c = (i < 8) ? (half * 8 + i) : (16 + half * 8 + (i - 8));
        b0[i] = (_Float16)Xg[(size_t)(c     ) * HWn];
        b1[i] = (_Float16)Xg[(size_t)(c + 32) * HWn];
    }

    v8f c0 = {}; v8f c1 = {};
    c0 = __builtin_amdgcn_wmma_f32_16x16x32_f16(false, a00, false, b0, (short)0, c0, false, false);
    c0 = __builtin_amdgcn_wmma_f32_16x16x32_f16(false, a01, false, b1, (short)0, c0, false, false);
    c1 = __builtin_amdgcn_wmma_f32_16x16x32_f16(false, a10, false, b0, (short)0, c1, false, false);
    c1 = __builtin_amdgcn_wmma_f32_16x16x32_f16(false, a11, false, b1, (short)0, c1, false, false);

    // ---- Epilogue: BN (inference) + ReLU + dot with w2, reduce over 32 chans.
    // C/D layout: lane l<16 -> VGPR v = channel v; lane l>=16 -> channel v+8.
    float partial = 0.f;
    #pragma unroll
    for (int v = 0; v < 8; ++v) {
        const int ch0 = half * 8 + v;        // from c0 (channels 0..15)
        const int ch1 = 16 + half * 8 + v;   // from c1 (channels 16..31)
        const float inv0 = bn_scale[ch0] * rsqrtf(bn_var[ch0] + 1e-5f);
        const float inv1 = bn_scale[ch1] * rsqrtf(bn_var[ch1] + 1e-5f);
        float h0 = c0[v] * inv0 + (bn_bias[ch0] - bn_mean[ch0] * inv0);
        float h1 = c1[v] * inv1 + (bn_bias[ch1] - bn_mean[ch1] * inv1);
        h0 = h0 > 0.f ? h0 : 0.f;
        h1 = h1 > 0.f ? h1 : 0.f;
        partial += w2[ch0] * h0 + w2[ch1] * h1;
    }
    partial += __shfl_xor(partial, 16, 32);  // combine lane / lane+16 halves
    if (half == 0)
        S[(size_t)g * HWn + q0 + lane] = partial;
}

// ---------------------------------------------------------------------------
// Kernel 2: per-pixel 3x3 reflect-pad covariance (K=8), diagonal loading,
// 8x8 Cholesky solve, Capon normalization.  One thread per pixel.
// ---------------------------------------------------------------------------
__global__ __launch_bounds__(256)
void capon_solve_kernel(const float* __restrict__ S,
                        const float* __restrict__ a0_raw,
                        const float* __restrict__ sigma_raw,
                        float* __restrict__ Wout)
{
    const int idx = blockIdx.x * blockDim.x + threadIdx.x;
    if (idx >= Bdim * HWn) return;
    const int b = idx / HWn;
    const int q = idx - b * HWn;
    const int y = q >> 7;
    const int x = q & (Wdim - 1);

    // softmax(a0_raw)
    float a0[8]; float mx = -1e30f;
    #pragma unroll
    for (int k = 0; k < 8; ++k) { a0[k] = a0_raw[k]; mx = fmaxf(mx, a0[k]); }
    float se = 0.f;
    #pragma unroll
    for (int k = 0; k < 8; ++k) { a0[k] = expf(a0[k] - mx); se += a0[k]; }
    const float inv_se = 1.f / se;
    #pragma unroll
    for (int k = 0; k < 8; ++k) a0[k] *= inv_se;

    // sigma = 0.3 * sigmoid(sigma_raw); loading term gamma * sigma^2
    float sig2[8];
    #pragma unroll
    for (int k = 0; k < 8; ++k) {
        const float sg = 0.3f / (1.f + expf(-sigma_raw[k]));
        sig2[k] = sg * sg;
    }

    // streaming 9-tap covariance accumulation
    float acc[8][8], mu[8];
    #pragma unroll
    for (int k = 0; k < 8; ++k) {
        mu[k] = 0.f;
        #pragma unroll
        for (int l = 0; l < 8; ++l) acc[k][l] = 0.f;
    }
    #pragma unroll
    for (int dy = -1; dy <= 1; ++dy) {
        int yy = y + dy;
        yy = (yy < 0) ? -yy : ((yy >= Hdim) ? (2 * Hdim - 2 - yy) : yy);
        #pragma unroll
        for (int dx = -1; dx <= 1; ++dx) {
            int xx = x + dx;
            xx = (xx < 0) ? -xx : ((xx >= Wdim) ? (2 * Wdim - 2 - xx) : xx);
            float s[8];
            #pragma unroll
            for (int k = 0; k < 8; ++k) {
                s[k] = S[(size_t)(b * Kstr + k) * HWn + yy * Wdim + xx];
                mu[k] += s[k];
            }
            #pragma unroll
            for (int k = 0; k < 8; ++k)
                #pragma unroll
                for (int l = 0; l <= k; ++l)
                    acc[k][l] += s[k] * s[l];
        }
    }

    // R_load = 0.9 R + (0.1 tr/8 + 1e-3) I + diag(sigma^2)
    const float invN = 1.f / 9.f;
    float M[8][8];
    float tr = 0.f;
    #pragma unroll
    for (int k = 0; k < 8; ++k) {
        #pragma unroll
        for (int l = 0; l <= k; ++l) {
            const float R = acc[k][l] * invN - (mu[k] * invN) * (mu[l] * invN);
            M[k][l] = 0.9f * R;
            M[l][k] = M[k][l];
            if (l == k) tr += R;
        }
    }
    const float loadv = 0.1f * tr * (1.f / 8.f) + 1e-3f;
    #pragma unroll
    for (int k = 0; k < 8; ++k) M[k][k] += loadv + sig2[k];

    // Cholesky (lower), fully unrolled
    #pragma unroll
    for (int j = 0; j < 8; ++j) {
        float d = M[j][j];
        #pragma unroll
        for (int t = 0; t < 8; ++t) if (t < j) d -= M[j][t] * M[j][t];
        const float sj = sqrtf(d);
        M[j][j] = sj;
        const float invs = 1.f / sj;
        #pragma unroll
        for (int i = 0; i < 8; ++i) {
            if (i > j) {
                float v = M[i][j];
                #pragma unroll
                for (int t = 0; t < 8; ++t) if (t < j) v -= M[i][t] * M[j][t];
                M[i][j] = v * invs;
            }
        }
    }
    // forward: L z = a0
    float z[8];
    #pragma unroll
    for (int i = 0; i < 8; ++i) {
        float v = a0[i];
        #pragma unroll
        for (int t = 0; t < 8; ++t) if (t < i) v -= M[i][t] * z[t];
        z[i] = v / M[i][i];
    }
    // backward: L^T u = z
    float u[8];
    #pragma unroll
    for (int ii = 0; ii < 8; ++ii) {
        const int i = 7 - ii;
        float v = z[i];
        #pragma unroll
        for (int t = 0; t < 8; ++t) if (t > i) v -= M[t][i] * u[t];
        u[i] = v / M[i][i];
    }
    float denom = 0.f;
    #pragma unroll
    for (int k = 0; k < 8; ++k) denom += u[k] * a0[k];
    const float invd = 1.f / denom;
    #pragma unroll
    for (int k = 0; k < 8; ++k)
        Wout[(size_t)idx * 8 + k] = u[k] * invd;
}

// ---------------------------------------------------------------------------
// Kernel 3: Y[b,c,h,w] = sum_k w[b,hw,k] * X[b,k,c,h,w]
// ---------------------------------------------------------------------------
__global__ __launch_bounds__(256)
void fuse_kernel(const float* __restrict__ X,
                 const float* __restrict__ Wt,
                 float* __restrict__ Y)
{
    const size_t idx = (size_t)blockIdx.x * blockDim.x + threadIdx.x;
    if (idx >= (size_t)Bdim * Cdim * HWn) return;
    const int b = (int)(idx / ((size_t)Cdim * HWn));
    const int r = (int)(idx - (size_t)b * Cdim * HWn);
    const int c = r / HWn;
    const int q = r - c * HWn;

    const float* wp = Wt + ((size_t)b * HWn + q) * 8;
    float acc = 0.f;
    #pragma unroll
    for (int k = 0; k < 8; ++k)
        acc += wp[k] * X[(((size_t)(b * Kstr + k)) * Cdim + c) * HWn + q];
    Y[idx] = acc;
}

// ---------------------------------------------------------------------------
extern "C" void kernel_launch(void* const* d_in, const int* in_sizes, int n_in,
                              void* d_out, int out_size, void* d_ws, size_t ws_size,
                              hipStream_t stream)
{
    const float* X         = (const float*)d_in[0];
    const float* w1        = (const float*)d_in[1];
    const float* bn_scale  = (const float*)d_in[2];
    const float* bn_bias   = (const float*)d_in[3];
    const float* bn_mean   = (const float*)d_in[4];
    const float* bn_var    = (const float*)d_in[5];
    const float* w2        = (const float*)d_in[6];
    const float* a0_raw    = (const float*)d_in[7];
    const float* sigma_raw = (const float*)d_in[8];
    float* Y = (float*)d_out;

    float* S  = (float*)d_ws;                       // [B*K, H*W]  = 1 MB
    float* Wc = S + (size_t)Bdim * Kstr * HWn;      // [B*H*W, K]  = 1 MB

    // Stage 1: sensor maps via WMMA (16384 wave-tiles, 8 waves/block)
    conv_sensor_kernel<<<2048, 256, 0, stream>>>(X, w1, bn_scale, bn_bias,
                                                 bn_mean, bn_var, w2, S);
    // Stage 2: per-pixel covariance + Capon solve
    const int P2 = Bdim * HWn;
    capon_solve_kernel<<<(P2 + 255) / 256, 256, 0, stream>>>(S, a0_raw, sigma_raw, Wc);
    // Stage 3: weighted fusion
    const size_t NY = (size_t)Bdim * Cdim * HWn;
    fuse_kernel<<<(unsigned)((NY + 255) / 256), 256, 0, stream>>>(X, Wc, Y);
}
